// ScaledDotProductAttention_17540646437188
// MI455X (gfx1250) — matrix operand order repaired and checked by reading, not compiled, not run on hardware
//
#include <hip/hip_runtime.h>
#include <hip/hip_bf16.h>

// ---- problem constants -----------------------------------------------------
#define BB   2
#define SS   2048
#define DD   768
#define HH   12
#define DH   64
// ws layout: Q | K | Vt, each B*H*S*DH bf16 (== B*S*D). V stored transposed
// [B,H,DH,S] so P@V B-operands are K-contiguous 32-bit loads.
#define MAT_ELEMS (BB * HH * SS * DH)

typedef __bf16 bf16_t;
typedef bf16_t v16bf __attribute__((ext_vector_type(16)));
typedef bf16_t v2bf  __attribute__((ext_vector_type(2)));
typedef float  v8f   __attribute__((ext_vector_type(8)));

union ABf16 { v16bf v; v2bf p[8]; unsigned u[8]; };

// native converts -> v_cvt_pk_bf16_f32 (1 VALU op per pair)
__device__ __forceinline__ v2bf cvt2(float lo, float hi) {
    v2bf r; r[0] = (bf16_t)lo; r[1] = (bf16_t)hi; return r;
}

// LDS score row stride: 2052 f32 => banks rotate by 4 per row (2048%64==0 bad)
#define SCW 2052

// ============================================================================
// Kernel 1: fused Q/K/V projections  Y = X @ W^T + b, stored bf16 split-head.
// One 16x16 output tile per wave; K-loop 768/32 = 24 WMMA bf16 ops (unroll 4).
// Q,K stored [B,H,S,DH]; V stored transposed [B,H,DH,S].
// ============================================================================
__global__ __launch_bounds__(256)
void proj_kernel(const float* __restrict__ q, const float* __restrict__ k,
                 const float* __restrict__ v,
                 const float* __restrict__ Wq, const float* __restrict__ bq,
                 const float* __restrict__ Wk, const float* __restrict__ bk,
                 const float* __restrict__ Wv, const float* __restrict__ bv,
                 bf16_t* __restrict__ ws)
{
    const int wave = threadIdx.x >> 5;
    const int lane = threadIdx.x & 31;
    const int half = lane >> 4;
    const int l16  = lane & 15;

    int tile = blockIdx.x * 8 + wave;                 // 0..36863
    int mat  = tile / (256 * 48);
    int rem  = tile % (256 * 48);
    int mt   = rem / 48;
    int nt   = rem % 48;

    const float* X    = (mat == 0) ? q  : (mat == 1) ? k  : v;
    const float* W    = (mat == 0) ? Wq : (mat == 1) ? Wk : Wv;
    const float* bias = (mat == 0) ? bq : (mat == 1) ? bk : bv;
    bf16_t* dst = ws + (size_t)mat * MAT_ELEMS;

    const int m = mt * 16 + l16;   // A-matrix row for this lane
    const int n = nt * 16 + l16;   // B-matrix col (== C col) for this lane
    const float* Xrow = X + (size_t)m * DD;
    const float* Wrow = W + (size_t)n * DD;

    v8f c = {};
#pragma unroll 4
    for (int kb = 0; kb < DD; kb += 32) {
        ABf16 a, bm;
#pragma unroll
        for (int j = 0; j < 8; ++j) {
            // A (16x32, 16-bit): VGPR j holds K = 16*(j/4) + 8*half + 2*(j%4)
            int ka = kb + ((j >> 2) << 4) + (half << 3) + ((j & 3) << 1);
            a.p[j] = cvt2(Xrow[ka], Xrow[ka + 1]);
            // B (32x16, 16-bit): VGPR j holds K = 16*(j/4) + 8*half + 2*(j%4) ; B[k][n]=W[n][k]
            int kbi = kb + ((j >> 2) << 4) + (half << 3) + ((j & 3) << 1);
            bm.p[j] = cvt2(Wrow[kbi], Wrow[kbi + 1]);
        }
        c = __builtin_amdgcn_wmma_f32_16x16x32_bf16(false, a.v, false, bm.v,
                                                    (short)0, c, false, false);
    }

    const float bbv = bias[n];
    const int hcol = n >> 6;       // head
    const int dh   = n & 63;       // within-head dim
#pragma unroll
    for (int r = 0; r < 8; ++r) {
        int gr = mt * 16 + r + 8 * half;        // global row in [0,4096)
        int bi = gr >> 11;                      // batch
        int si = gr & 2047;                     // sequence
        size_t idx = (mat == 2)
            ? ((size_t)(bi * HH + hcol) * DH + dh) * SS + si      // V transposed
            : (((size_t)(bi * HH + hcol) * SS) + si) * DH + dh;   // Q/K row-major
        dst[idx] = (bf16_t)(c[r] + bbv);
    }
}

// ============================================================================
// Kernel 2: attention for one (b, h, 16-row q strip).
//   Phase 1: scores = Q K^T into LDS (raw)          -- 2 bf16 WMMA / k-tile
//   Phase 2: scale+mask+softmax in LDS (float2 / b64 DS ops), stream
//            normalized probs to d_out once; LDS keeps UNNORMALIZED e
//   Phase 3: ctx = (e @ V) * inv_row  -- normalization folded into epilogue
// LDS: 16*2052 f32 scores (~128KB, bank-padded) + ~5KB  (CDNA5 320KB WGP LDS)
// ============================================================================
__global__ __launch_bounds__(256)
void attn_kernel(const bf16_t* __restrict__ ws,
                 const unsigned char* __restrict__ mask,
                 float* __restrict__ out)
{
    __shared__ float sc[16 * SCW];
    __shared__ float red[16][16];
    __shared__ float rowv[16];     // row max, then 1/rowsum
    __shared__ float ps[4][16][16];

    const bf16_t* Qw = ws;
    const bf16_t* Kw = ws + (size_t)MAT_ELEMS;
    const bf16_t* Vw = ws + (size_t)2 * MAT_ELEMS;

    const int qt = blockIdx.x & 127;       // 128 q-strips of 16 rows
    const int bh = blockIdx.x >> 7;        // 0..23
    const int b  = bh / HH;
    const int h  = bh % HH;
    const int q0 = qt * 16;

    const int tid  = threadIdx.x;
    const int wave = tid >> 5;
    const int lane = tid & 31;
    const int half = lane >> 4;
    const int l16  = lane & 15;

    const bf16_t* Qbase = Qw + (((size_t)(b * HH + h) * SS) + q0) * DH;
    const bf16_t* Kbase = Kw + ((size_t)(b * HH + h) * SS) * DH;
    const bf16_t* Vt    = Vw + ((size_t)(b * HH + h) * DH) * SS;  // [DH][S]

    // ---------------- Phase 1: raw scores into LDS --------------------------
    ABf16 a0, a1;                            // Q tile for kb=0 and kb=32
#pragma unroll
    for (int j = 0; j < 8; ++j) {
        int ka = ((j >> 2) << 4) + (half << 3) + ((j & 3) << 1);
        a0.u[j] = *(const unsigned*)(Qbase + (size_t)l16 * DH + ka);
        a1.u[j] = *(const unsigned*)(Qbase + (size_t)l16 * DH + 32 + ka);
    }
    for (int kt = wave; kt < 128; kt += 8) {
        const int ncol = kt * 16 + l16;      // key position (B col == C col)
        if (kt + 8 < 128)                    // gfx1250 global_prefetch_b8
            __builtin_prefetch(Kbase + (size_t)(ncol + 128) * DH, 0, 0);
        ABf16 b0, b1;
#pragma unroll
        for (int j = 0; j < 8; ++j) {
            int kk = ((j >> 2) << 4) + (half << 3) + ((j & 3) << 1); // B[k][n] = K[n][k] (row-major K)
            b0.u[j] = *(const unsigned*)(Kbase + (size_t)ncol * DH + kk);
            b1.u[j] = *(const unsigned*)(Kbase + (size_t)ncol * DH + 32 + kk);
        }
        v8f c = {};
        c = __builtin_amdgcn_wmma_f32_16x16x32_bf16(false, a0.v, false, b0.v,
                                                    (short)0, c, false, false);
        c = __builtin_amdgcn_wmma_f32_16x16x32_bf16(false, a1.v, false, b1.v,
                                                    (short)0, c, false, false);
#pragma unroll
        for (int r = 0; r < 8; ++r)
            sc[(r + 8 * half) * SCW + ncol] = c[r];
    }
    __syncthreads();

    // ---------------- Phase 2: scale + mask + softmax(e) + probs out --------
    // 16 threads per row; each owns float2 pairs at cc = 2*(sub + 16*i):
    //  - b64 DS ops, 16 threads of a row cover 32 distinct banks
    //  - probs writes: 8B/thread -> 128B contiguous bursts per row group
    const int row = tid >> 4;
    const int sub = tid & 15;
    const int qg  = q0 + row;
    const unsigned char* mrow = mask + ((size_t)b * SS + qg) * SS;
    float* scrow = &sc[row * SCW];
    __builtin_prefetch(mrow + sub * 128, 0, 0);

    float mx = -3.4e38f;
#pragma unroll 4
    for (int i = 0; i < 64; ++i) {
        int cc = (sub + (i << 4)) << 1;
        float2 s2 = *(float2*)(scrow + cc);
        unsigned short mm = *(const unsigned short*)(mrow + cc);
        float sa = s2.x * 0.125f; if (mm & 0xff) sa = -1e9f;   // 1/sqrt(64)
        float sb = s2.y * 0.125f; if (mm >> 8)   sb = -1e9f;
        *(float2*)(scrow + cc) = make_float2(sa, sb);
        mx = fmaxf(mx, fmaxf(sa, sb));
    }
    red[row][sub] = mx;
    __syncthreads();
    if (sub == 0) {
        float m2 = red[row][0];
        for (int i = 1; i < 16; ++i) m2 = fmaxf(m2, red[row][i]);
        rowv[row] = m2;
    }
    __syncthreads();
    const float rm = rowv[row];
    float sum = 0.f;
#pragma unroll 4
    for (int i = 0; i < 64; ++i) {
        int cc = (sub + (i << 4)) << 1;
        float2 s2 = *(float2*)(scrow + cc);
        float ea = __expf(s2.x - rm);
        float eb = __expf(s2.y - rm);
        *(float2*)(scrow + cc) = make_float2(ea, eb);  // keep UNNORMALIZED e
        sum += ea + eb;
    }
    red[row][sub] = sum;
    __syncthreads();
    if (sub == 0) {
        float s2 = 0.f;
        for (int i = 0; i < 16; ++i) s2 += red[row][i];
        rowv[row] = 1.0f / s2;                          // row inv
    }
    __syncthreads();
    const float inv = rowv[row];
    float* probs = out + (size_t)BB * SS * DD
                       + (((size_t)(b * HH + h) * SS) + qg) * SS;
#pragma unroll 4
    for (int i = 0; i < 64; ++i) {
        int cc = (sub + (i << 4)) << 1;
        float2 e2 = *(float2*)(scrow + cc);
        *(float2*)(probs + cc) = make_float2(e2.x * inv, e2.y * inv);
    }

    // ---------------- Phase 3: ctx = (e @ V) * inv --------------------------
    // Normalization folded into the epilogue: ctx[q,:] = inv_q * sum_k e_qk V_k
    const int nt    = wave & 3;              // dh tile (4 x 16 = 64)
    const int khalf = wave >> 2;             // split K range across wave pairs
    const int ncol  = nt * 16 + l16;
    const bf16_t* Vcol = Vt + (size_t)ncol * SS;   // K-contiguous column

    v8f cc = {};
#pragma unroll 2
    for (int kb = khalf * 1024; kb < khalf * 1024 + 1024; kb += 32) {
        ABf16 a, bm;
#pragma unroll
        for (int j = 0; j < 8; ++j) {
            int ka = kb + ((j >> 2) << 4) + (half << 3) + ((j & 3) << 1);
            float2 t = *(const float2*)(&sc[l16 * SCW + ka]);  // one b64 DS read
            a.p[j] = cvt2(t.x, t.y);
            int kr = kb + ((j >> 2) << 4) + (half << 3) + ((j & 3) << 1);     // B[k][n] = Vt[n][k]
            bm.u[j] = *(const unsigned*)(Vcol + kr);  // one aligned b32 load
        }
        cc = __builtin_amdgcn_wmma_f32_16x16x32_bf16(false, a.v, false, bm.v,
                                                     (short)0, cc, false, false);
    }
    if (khalf == 1) {
#pragma unroll
        for (int r = 0; r < 8; ++r) ps[nt][r + 8 * half][l16] = cc[r];
    }
    __syncthreads();
    if (khalf == 0) {
#pragma unroll
        for (int r = 0; r < 8; ++r) {
            int mrow2 = r + 8 * half;
            float val = (cc[r] + ps[nt][mrow2][l16]) * rowv[mrow2];
            int qq = q0 + mrow2;
            out[((size_t)b * SS + qq) * DD + h * DH + nt * 16 + l16] = val;
        }
    }
}

// ============================================================================
extern "C" void kernel_launch(void* const* d_in, const int* in_sizes, int n_in,
                              void* d_out, int out_size, void* d_ws, size_t ws_size,
                              hipStream_t stream) {
    (void)in_sizes; (void)n_in; (void)out_size; (void)ws_size;
    const float* q  = (const float*)d_in[0];
    const float* k  = (const float*)d_in[1];
    const float* v  = (const float*)d_in[2];
    const float* Wq = (const float*)d_in[3];
    const float* bq = (const float*)d_in[4];
    const float* Wk = (const float*)d_in[5];
    const float* bk = (const float*)d_in[6];
    const float* Wv = (const float*)d_in[7];
    const float* bv = (const float*)d_in[8];
    const unsigned char* mask = (const unsigned char*)d_in[9];

    bf16_t* ws = (bf16_t*)d_ws;        // Q|K|Vt bf16, 3 * 6.29 MB
    float* out = (float*)d_out;

    // 3 mats * 256 * 48 tiles / 8 waves per block = 4608 blocks
    proj_kernel<<<4608, 256, 0, stream>>>(q, k, v, Wq, bq, Wk, bk, Wv, bv, ws);
    // B * H * (S/16) = 2 * 12 * 128 = 3072 blocks
    attn_kernel<<<3072, 256, 0, stream>>>(ws, mask, out);
}
